// NeighbourApproxPCA_36094905155928
// MI455X (gfx1250) — matrix operand order repaired and checked by reading, not compiled, not run on hardware
//
#include <hip/hip_runtime.h>

typedef __attribute__((ext_vector_type(16))) _Float16 v16h;
typedef __attribute__((ext_vector_type(8)))  float    v8f;

#define EPSV 1e-6f

__device__ __forceinline__ float eluf(float x) {
    return x > 0.0f ? x : (__expf(x) - 1.0f);
}

// sum across lane pairs (L, L^16) via LDS swizzle SWAPX16 (xor=0x10, and=0x1f)
__device__ __forceinline__ float xsum16(float x) {
    int y = __builtin_amdgcn_ds_swizzle(__float_as_int(x), 0x401F);
    return x + __int_as_float(y);
}

__global__ __launch_bounds__(256)
void neighbour_pca_kernel(const float* __restrict__ coords,
                          const float* __restrict__ distsq,
                          const float* __restrict__ feats,
                          const int*   __restrict__ nidx,
                          const float* __restrict__ W0, const float* __restrict__ b0,
                          const float* __restrict__ W1, const float* __restrict__ b1,
                          const float* __restrict__ W2, const float* __restrict__ b2,
                          float* __restrict__ out, int V)
{
    // Weights pre-packed into WMMA B-fragment layout (f16):
    //   B elem j at lane L holds W[K][N], K = (j&7) + 8*(L/16) + 16*(j/8), N = 16*tile + L%16
    __shared__ __align__(32) _Float16 Bp[3][2][32][16];
    __shared__ float biasS[3][32];
    // Per-wave staging: next-layer A fragments, lane L reads Ap[wave][L][0..15]
    __shared__ __align__(32) _Float16 Ap[8][32][16];

    const int tid = threadIdx.x;
    for (int i = tid; i < 1024; i += 256) {
        int tile = i >> 9;
        int ln   = (i >> 4) & 31;
        int j    = i & 15;
        int K = (j & 7) + ((ln >> 4) << 3) + ((j >> 3) << 4);
        int N = (tile << 4) + (ln & 15);
        float w0v = (K < 9) ? W0[K * 32 + N] : 0.0f;   // pad K: 9 -> 32
        Bp[0][tile][ln][j] = (_Float16)w0v;
        Bp[1][tile][ln][j] = (_Float16)W1[K * 32 + N];
        Bp[2][tile][ln][j] = (_Float16)W2[K * 32 + N];
    }
    if (tid < 32) {
        biasS[0][tid] = b0[tid];
        biasS[1][tid] = b1[tid];
        biasS[2][tid] = b2[tid];
    }
    __syncthreads();

    const int wave = tid >> 5;
    const int lane = tid & 31;
    const int v = blockIdx.x * 8 + wave;
    if (v >= V) return;                 // wave-uniform

    // ---------- Phase A: weighted mean + covariance ----------
    const int f = lane & 15;            // feature handled by this lane
    const int kbase = (lane >> 4) << 4; // lanes 0-15: k=0..15, lanes 16-31: k=16..31
    float wsum = 0.f;
    float s1x = 0.f, s1y = 0.f, s1z = 0.f;
    float sxx = 0.f, sxy = 0.f, sxz = 0.f, syy = 0.f, syz = 0.f, szz = 0.f;
    #pragma unroll 4
    for (int kk = 0; kk < 16; ++kk) {
        int k = kbase + kk;
        int idx = nidx[v * 32 + k];
        float d = distsq[v * 32 + k];
        bool valid = idx >= 0;
        int ic = valid ? idx : 0;
        float e = valid ? __expf(-10.0f * d) : 0.0f;
        float w = feats[ic * 16 + f] * e;
        float cx = coords[ic * 3 + 0];
        float cy = coords[ic * 3 + 1];
        float cz = coords[ic * 3 + 2];
        wsum += w;
        s1x += w * cx; s1y += w * cy; s1z += w * cz;
        sxx += w * cx * cx; sxy += w * cx * cy; sxz += w * cx * cz;
        syy += w * cy * cy; syz += w * cy * cz; szz += w * cz * cz;
    }
    wsum = xsum16(wsum);
    s1x = xsum16(s1x); s1y = xsum16(s1y); s1z = xsum16(s1z);
    sxx = xsum16(sxx); sxy = xsum16(sxy); sxz = xsum16(sxz);
    syy = xsum16(syy); syz = xsum16(syz); szz = xsum16(szz);

    float inv = 1.0f / (wsum + EPSV);
    float mx = s1x * inv, my = s1y * inv, mz = s1z * inv;
    float c9[9];
    c9[0] = sxx * inv - mx * mx; c9[1] = sxy * inv - mx * my; c9[2] = sxz * inv - mx * mz;
    c9[3] = c9[1];               c9[4] = syy * inv - my * my; c9[5] = syz * inv - my * mz;
    c9[6] = c9[2];               c9[7] = c9[5];               c9[8] = szz * inv - mz * mz;

    float* outv = out + (size_t)v * 560;
    if (lane < 16) {                    // means -> out[v, 512 + f*3 + c]
        outv[512 + f * 3 + 0] = mx;
        outv[512 + f * 3 + 1] = my;
        outv[512 + f * 3 + 2] = mz;
    }

    // ---------- Layer-0 A fragment (16-bit A layout, K padded 9->32) ----------
    // elem j at lane L holds A[M=L%16][K], K = (j&7) + 8*(L/16) + 16*(j/8)
    // lanes 0-15: j=0..7 -> K=0..7 ; lanes 16-31: j=0 -> K=8 ; everything else 0.
    bool lo = lane < 16;
    v16h a;
    #pragma unroll
    for (int j = 0; j < 16; ++j) a[j] = (_Float16)0.0f;
    a[0] = (_Float16)(lo ? c9[0] : c9[8]);
    #pragma unroll
    for (int j = 1; j < 8; ++j) a[j] = (_Float16)(lo ? c9[j] : 0.0f);

    // ---------- 3-layer MLP via WMMA (N=32 -> two 16x16 tiles per layer) ----------
    v8f d0, d1;
    #pragma unroll
    for (int l = 0; l < 3; ++l) {
        v16h bf0 = *(const v16h*)(&Bp[l][0][lane][0]);
        v16h bf1 = *(const v16h*)(&Bp[l][1][lane][0]);
        v8f z = {};
        d0 = __builtin_amdgcn_wmma_f32_16x16x32_f16(false, a, false, bf0,
                                                    (short)0, z, false, false);
        d1 = __builtin_amdgcn_wmma_f32_16x16x32_f16(false, a, false, bf1,
                                                    (short)0, z, false, false);
        float bA = biasS[l][lane & 15];
        float bB = biasS[l][16 + (lane & 15)];
        #pragma unroll
        for (int r = 0; r < 8; ++r) {
            d0[r] = eluf(d0[r] + bA);
            d1[r] = eluf(d1[r] + bB);
        }
        if (l < 2) {
            // D (C/D layout) -> A layout staging in LDS.
            // D elem r at lane L is h[M][N], M = r + 8*(L/16), N = 16*tile + L%16.
            // A-layout slot: row La = M + 16*((N%16)/8), elem j = (N%8) + 8*(N/16).
            int half = ((lane & 15) >> 3) << 4;      // 0 or 16
            int jlo  = lane & 7;
            #pragma unroll
            for (int r = 0; r < 8; ++r) {
                int M = r + ((lane >> 4) << 3);
                Ap[wave][M + half][jlo]     = (_Float16)d0[r];  // tile 0: j = N%8
                Ap[wave][M + half][jlo + 8] = (_Float16)d1[r];  // tile 1: j = N%8 + 8
            }
            asm volatile("s_wait_dscnt 0" ::: "memory");
            __builtin_amdgcn_wave_barrier();
            a = *(const v16h*)(&Ap[wave][lane][0]);
        }
    }

    // ---------- Final h -> out[v, f*32 + n] ----------
    #pragma unroll
    for (int r = 0; r < 8; ++r) {
        int M = r + ((lane >> 4) << 3);
        outv[M * 32 + (lane & 15)]      = d0[r];
        outv[M * 32 + 16 + (lane & 15)] = d1[r];
    }
}

extern "C" void kernel_launch(void* const* d_in, const int* in_sizes, int n_in,
                              void* d_out, int out_size, void* d_ws, size_t ws_size,
                              hipStream_t stream) {
    const float* coords = (const float*)d_in[0];
    const float* distsq = (const float*)d_in[1];
    const float* feats  = (const float*)d_in[2];
    const int*   nidx   = (const int*)d_in[3];
    const float* W0 = (const float*)d_in[4];
    const float* b0 = (const float*)d_in[5];
    const float* W1 = (const float*)d_in[6];
    const float* b1 = (const float*)d_in[7];
    const float* W2 = (const float*)d_in[8];
    const float* b2 = (const float*)d_in[9];
    float* out = (float*)d_out;

    int V = in_sizes[0] / 3;
    int blocks = (V + 7) / 8;   // 8 waves per block, one point per wave
    neighbour_pca_kernel<<<blocks, 256, 0, stream>>>(
        coords, distsq, feats, nidx, W0, b0, W1, b1, W2, b2, out, V);
}
